// CausalSelfAttention_9405978378751
// MI455X (gfx1250) — compile-verified, matrix operand order checked by reading, not run on hardware
//
#include <hip/hip_runtime.h>
#include <hip/hip_bf16.h>

// ---------------------------------------------------------------------------
// CDNA5 (gfx1250) causal self-attention:
//   - all matmuls on v_wmma_f32_16x16x32_bf16
//   - V tiles staged global->LDS via global_load_async_to_lds_b128 (ASYNCcnt)
//   - K tiles prefetched with global_prefetch_b8
// ---------------------------------------------------------------------------

typedef __bf16 bf16_t;
typedef __attribute__((ext_vector_type(16))) __bf16 v16bf;
typedef __attribute__((ext_vector_type(8)))  float  v8f;
typedef __attribute__((ext_vector_type(4)))  unsigned int v4u;

#define VSTR 72   // LDS V-tile row stride in elements (144 B: 16B-aligned, bank-spread)

union FragU { v16bf v; v4u u[2]; };

__device__ __forceinline__ v8f v8f_zero() {
    v8f z;
#pragma unroll
    for (int i = 0; i < 8; ++i) z[i] = 0.0f;
    return z;
}

__device__ __forceinline__ v8f wmma_bf16(v16bf a, v16bf b, v8f c) {
    // D = A(16x32 bf16) * B(32x16 bf16) + C(16x16 f32)
    return __builtin_amdgcn_wmma_f32_16x16x32_bf16(
        /*neg_a=*/false, a, /*neg_b=*/false, b,
        /*c_mod=*/(short)0, c, /*reuse_a=*/false, /*reuse_b=*/false);
}

// Load a 16x32 bf16 A-fragment (or B-fragment = W^T with rows = N) from a
// row-major matrix. Per the CDNA5 VGPR layout, lanes 0-15 hold K 0-7 & 16-23,
// lanes 16-31 hold K 8-15 & 24-31 -> two contiguous 16-byte loads per lane.
__device__ __forceinline__ v16bf load_frag16(const bf16_t* __restrict__ base,
                                             int row0, int k0, int ld) {
    const int lane = threadIdx.x & 31;
    const bf16_t* p = base + (size_t)(row0 + (lane & 15)) * ld
                           + k0 + ((lane >> 4) << 3);
    FragU f;
    f.u[0] = *(const v4u*)(p);
    f.u[1] = *(const v4u*)(p + 16);
    return f.v;
}

// Gather a 32x16 bf16 B-fragment (B(k,n) = tile[k][n]) from an LDS-staged
// row-major tile with row stride VSTR.
__device__ __forceinline__ v16bf load_vfrag_lds(const bf16_t* V, int n0) {
    const int lane = threadIdx.x & 31;
    const int n    = n0 + (lane & 15);
    const int hi8  = (lane >> 4) << 3;
    v16bf r;
#pragma unroll
    for (int i = 0; i < 8; ++i) {
        r[i]     = V[(hi8 + i) * VSTR + n];
        r[i + 8] = V[(16 + hi8 + i) * VSTR + n];
    }
    return r;
}

__device__ __forceinline__ float redmax16(float v) {
#pragma unroll
    for (int m = 1; m < 16; m <<= 1) v = fmaxf(v, __shfl_xor(v, m, 32));
    return v;
}
__device__ __forceinline__ float redsum16(float v) {
#pragma unroll
    for (int m = 1; m < 16; m <<= 1) v += __shfl_xor(v, m, 32);
    return v;
}

// ---------------------------------------------------------------------------
// fp32 -> bf16 conversion
// ---------------------------------------------------------------------------
__global__ void f32_to_bf16_kernel(const float* __restrict__ s,
                                   bf16_t* __restrict__ d, int n) {
    int i      = blockIdx.x * blockDim.x + threadIdx.x;
    int stride = gridDim.x * blockDim.x;
    for (; i < n; i += stride) d[i] = (bf16_t)s[i];
}

// ---------------------------------------------------------------------------
// C[M,N] = A[M,K] @ W[N,K]^T   (all bf16 inputs, f32 accumulate)
// Block = 128 threads = 4 waves (2x2). Block tile 64x128, wave tile 32x64.
// ---------------------------------------------------------------------------
template <int BF16OUT>
__global__ void __launch_bounds__(128)
gemm_nt_kernel(const bf16_t* __restrict__ A, const bf16_t* __restrict__ W,
               void* __restrict__ Cout, int M, int N, int K) {
    const int wid  = threadIdx.x >> 5;
    const int lane = threadIdx.x & 31;
    const int m0   = blockIdx.x * 64  + (wid & 1) * 32;
    const int n0   = blockIdx.y * 128 + (wid >> 1) * 64;

    v8f acc[2][4];
#pragma unroll
    for (int mi = 0; mi < 2; ++mi)
#pragma unroll
        for (int nf = 0; nf < 4; ++nf) acc[mi][nf] = v8f_zero();

    for (int k0 = 0; k0 < K; k0 += 32) {
        v16bf a[2], b[4];
        a[0] = load_frag16(A, m0,      k0, K);
        a[1] = load_frag16(A, m0 + 16, k0, K);
#pragma unroll
        for (int nf = 0; nf < 4; ++nf)
            b[nf] = load_frag16(W, n0 + nf * 16, k0, K);
#pragma unroll
        for (int mi = 0; mi < 2; ++mi)
#pragma unroll
            for (int nf = 0; nf < 4; ++nf)
                acc[mi][nf] = wmma_bf16(a[mi], b[nf], acc[mi][nf]);
    }

    const int hi = lane >> 4, c = lane & 15;
#pragma unroll
    for (int mi = 0; mi < 2; ++mi)
#pragma unroll
        for (int nf = 0; nf < 4; ++nf)
#pragma unroll
            for (int v = 0; v < 8; ++v) {
                const size_t r  = m0 + mi * 16 + v + 8 * hi;
                const size_t cc = n0 + nf * 16 + c;
                if (BF16OUT)
                    ((bf16_t*)Cout)[r * N + cc] = (bf16_t)acc[mi][nf][v];
                else
                    ((float*)Cout)[r * N + cc] = acc[mi][nf][v];
            }
}

// ---------------------------------------------------------------------------
// Flash attention: one wave per (head, 32-row query tile). qkv is bf16
// [4096, 3072] = [T, q|k|v]; output y is bf16 [4096, 1024].
// ---------------------------------------------------------------------------
__global__ void __launch_bounds__(128)
attn_kernel(const bf16_t* __restrict__ qkv, bf16_t* __restrict__ y) {
    const int lane = threadIdx.x & 31;
    const int wid  = threadIdx.x >> 5;
    const int gw   = blockIdx.x * 4 + wid;     // 0 .. 2047
    const int h    = gw >> 7;                  // 16 heads
    const int q0   = (gw & 127) * 32;          // 128 query tiles of 32 rows

    __shared__ bf16_t plds[4][32 * 40];        // per-wave P staging (padded)
    __shared__ bf16_t vlds[4][32 * VSTR];      // per-wave async-staged V tile
    bf16_t* myp = &plds[wid][0];
    bf16_t* myv = &vlds[wid][0];
    const unsigned vbase = (unsigned)(size_t)(void*)myv;  // LDS byte offset

    const int ld = 3072;
    const bf16_t* Qp = qkv + h * 64;
    const bf16_t* Kp = qkv + 1024 + h * 64;
    const bf16_t* Vp = qkv + 2048 + h * 64;

    // Q fragments: 2 M-tiles x 2 K-steps (hd = 64)
    v16bf qf[2][2];
#pragma unroll
    for (int mi = 0; mi < 2; ++mi)
#pragma unroll
        for (int ks = 0; ks < 2; ++ks)
            qf[mi][ks] = load_frag16(Qp, q0 + mi * 16, ks * 32, ld);

    float mrow[2][8], lrow[2][8];
    v8f   acc[2][4];
#pragma unroll
    for (int mi = 0; mi < 2; ++mi) {
#pragma unroll
        for (int v = 0; v < 8; ++v) { mrow[mi][v] = -1e30f; lrow[mi][v] = 0.0f; }
#pragma unroll
        for (int nf = 0; nf < 4; ++nf) acc[mi][nf] = v8f_zero();
    }

    const float sc = 0.125f * 1.44269504088896f;   // 1/sqrt(64) * log2(e)
    const int   hi = lane >> 4, cc = lane & 15;

    // async-staging lane roles: 4 rows x 8 chunks of 16 B per instruction
    const int vrow = lane >> 3;   // 0..3
    const int vch  = lane & 7;    // 0..7

    for (int j0 = 0; j0 <= q0; j0 += 32) {
        // ---- kick off async DMA of the V tile (32 x 64 bf16) into LDS ----
        // Overlaps with the QK^T WMMAs and the softmax below (ASYNCcnt).
#pragma unroll
        for (int i = 0; i < 8; ++i) {
            const int row = i * 4 + vrow;
            const bf16_t* g = Vp + (size_t)(j0 + row) * ld + vch * 8;
            const unsigned l = vbase + (unsigned)(row * (VSTR * 2) + vch * 16);
            asm volatile("global_load_async_to_lds_b128 %0, %1, off"
                         :: "v"(l), "v"(g) : "memory");
        }

        // ---- prefetch next iteration's K tile into cache ----
        if (j0 + 32 <= q0)
            __builtin_prefetch(Kp + (size_t)(j0 + 32 + lane) * ld, 0, 3);

        // ---- S = Q @ K^T for a 32x32 tile ----
        v16bf kb[2][2];
#pragma unroll
        for (int nf = 0; nf < 2; ++nf)
#pragma unroll
            for (int ks = 0; ks < 2; ++ks)
                kb[nf][ks] = load_frag16(Kp, j0 + nf * 16, ks * 32, ld);

        v8f s[2][2];
#pragma unroll
        for (int mi = 0; mi < 2; ++mi)
#pragma unroll
            for (int nf = 0; nf < 2; ++nf)
                s[mi][nf] = wmma_bf16(qf[mi][1], kb[nf][1],
                            wmma_bf16(qf[mi][0], kb[nf][0], v8f_zero()));

        const bool diag = (j0 == q0);
#pragma unroll
        for (int mi = 0; mi < 2; ++mi)
#pragma unroll
            for (int nf = 0; nf < 2; ++nf)
#pragma unroll
                for (int v = 0; v < 8; ++v) {
                    float val = s[mi][nf][v] * sc;
                    if (diag && (nf * 16 + cc > mi * 16 + v + 8 * hi))
                        val = -1e30f;              // causal mask (diag tile only)
                    s[mi][nf][v] = val;
                }

        // ---- online softmax (row stats live in C-fragment layout) ----
#pragma unroll
        for (int mi = 0; mi < 2; ++mi)
#pragma unroll
            for (int v = 0; v < 8; ++v) {
                float rm = redmax16(fmaxf(s[mi][0][v], s[mi][1][v]));
                float mn = fmaxf(mrow[mi][v], rm);
                float f  = exp2f(mrow[mi][v] - mn);
                mrow[mi][v] = mn;
                float rs = 0.0f;
#pragma unroll
                for (int nf = 0; nf < 2; ++nf) {
                    float p = exp2f(s[mi][nf][v] - mn);
                    s[mi][nf][v] = p;
                    rs += p;
                }
                rs = redsum16(rs);
                lrow[mi][v] = lrow[mi][v] * f + rs;
#pragma unroll
                for (int nf2 = 0; nf2 < 4; ++nf2) acc[mi][nf2][v] *= f;
            }

        // ---- C-layout -> A-layout via LDS round trip ----
#pragma unroll
        for (int mi = 0; mi < 2; ++mi)
#pragma unroll
            for (int nf = 0; nf < 2; ++nf)
#pragma unroll
                for (int v = 0; v < 8; ++v)
                    myp[(mi * 16 + v + 8 * hi) * 40 + nf * 16 + cc] =
                        (bf16_t)s[mi][nf][v];
        asm volatile("s_wait_dscnt 0" ::: "memory");   // wave-local LDS ordering

        v16bf pa[2];
        pa[0] = load_frag16(myp, 0,  0, 40);
        pa[1] = load_frag16(myp, 16, 0, 40);

        // ---- wait for the async V DMA, then acc += P @ V from LDS ----
        asm volatile("s_wait_asynccnt 0x0" ::: "memory");
#pragma unroll
        for (int nf2 = 0; nf2 < 4; ++nf2) {
            v16bf vb = load_vfrag_lds(myv, nf2 * 16);
            acc[0][nf2] = wmma_bf16(pa[0], vb, acc[0][nf2]);
            acc[1][nf2] = wmma_bf16(pa[1], vb, acc[1][nf2]);
        }
        // NOTE: the s_wait_dscnt implied before the PV WMMAs (register deps on
        // the ds gathers) retires all LDS reads before the next iteration's
        // async stores can be issued, so the single V buffer is race-free.
    }

    // ---- normalize and store y ----
#pragma unroll
    for (int mi = 0; mi < 2; ++mi)
#pragma unroll
        for (int nf2 = 0; nf2 < 4; ++nf2)
#pragma unroll
            for (int v = 0; v < 8; ++v) {
                const size_t t = q0 + mi * 16 + v + 8 * hi;
                y[t * 1024 + h * 64 + nf2 * 16 + cc] =
                    (bf16_t)(acc[mi][nf2][v] / lrow[mi][v]);
            }
}

// ---------------------------------------------------------------------------
// Host-side orchestration
// ---------------------------------------------------------------------------
extern "C" void kernel_launch(void* const* d_in, const int* in_sizes, int n_in,
                              void* d_out, int out_size, void* d_ws, size_t ws_size,
                              hipStream_t stream) {
    (void)in_sizes; (void)n_in; (void)out_size; (void)ws_size;
    const float* x  = (const float*)d_in[0];   // [1,4096,1024]
    const float* wa = (const float*)d_in[1];   // [3072,1024]
    const float* wp = (const float*)d_in[2];   // [1024,1024]
    float* out = (float*)d_out;                // [1,4096,1024]

    const int T = 4096, C = 1024, C3 = 3072;
    char* ws = (char*)d_ws;
    // Workspace layout (yb overlays xb; x is dead after the QKV GEMM):
    bf16_t* xb  = (bf16_t*)(ws);                          //  8 MB
    bf16_t* wab = (bf16_t*)(ws +  8ull * 1024 * 1024);    //  6 MB
    bf16_t* wpb = (bf16_t*)(ws + 14ull * 1024 * 1024);    //  2 MB
    bf16_t* qkv = (bf16_t*)(ws + 16ull * 1024 * 1024);    // 24 MB
    bf16_t* yb  = (bf16_t*)(ws);                          //  8 MB (reuse)

    // 1) fp32 -> bf16 staging
    f32_to_bf16_kernel<<<2048, 256, 0, stream>>>(x,  xb,  T * C);
    f32_to_bf16_kernel<<<2048, 256, 0, stream>>>(wa, wab, C3 * C);
    f32_to_bf16_kernel<<<1024, 256, 0, stream>>>(wp, wpb, C * C);

    // 2) qkv = x @ w_attn^T   [4096, 3072] bf16
    gemm_nt_kernel<1><<<dim3(T / 64, C3 / 128), 128, 0, stream>>>(
        xb, wab, (void*)qkv, T, C3, C);

    // 3) flash attention -> y [4096, 1024] bf16
    attn_kernel<<<(16 * (T / 32)) / 4, 128, 0, stream>>>(qkv, yb);

    // 4) out = y @ w_proj^T   [4096, 1024] f32
    gemm_nt_kernel<0><<<dim3(T / 64, C / 128), 128, 0, stream>>>(
        yb, wpb, (void*)out, T, C, C);
}
